// HeteroGATLayer_20177756356843
// MI455X (gfx1250) — matrix operand order repaired and checked by reading, not compiled, not run on hardware
//
#include <hip/hip_runtime.h>
#include <math.h>

// ---------------------------------------------------------------- types
typedef __attribute__((ext_vector_type(16))) __bf16          bf16x16;
typedef __attribute__((ext_vector_type(16))) unsigned short  ushort16;
typedef __attribute__((ext_vector_type(8)))  float           floatx8;
typedef __attribute__((ext_vector_type(4)))  float           float4v;
typedef __attribute__((ext_vector_type(4)))  unsigned int    uint4v;
typedef __attribute__((ext_vector_type(2)))  unsigned int    uint2v;

union FragBF { ushort16 u; bf16x16 b; uint4v q[2]; };

// fp32 -> bf16, round-to-nearest-even
__device__ inline unsigned short f2bf(float f) {
  unsigned u = __float_as_uint(f);
  u += 0x7FFFu + ((u >> 16) & 1u);
  return (unsigned short)(u >> 16);
}
__device__ inline unsigned int pack2(float lo, float hi) {
  return (unsigned)f2bf(lo) | ((unsigned)f2bf(hi) << 16);
}

// ============================================================= WMMA GEMM
// Y[m,0:128] = X[m,0:K] @ W[0:K,0:128] (+bias). One block (8 waves, 256thr)
// per 128-row strip; A fragment loaded once per k-step, reused by 8 WMMAs.
// Staging: b128 global loads, bf16-packed b64 LDS stores for X.
__global__ __launch_bounds__(256) void gemm_wmma(
    const float* __restrict__ X, const float* __restrict__ W,
    const float* __restrict__ bias, float* __restrict__ Y,
    int K, int ldx, int ldw, int ldy)
{
  __shared__ unsigned short Xlds[128 * 32];
  __shared__ unsigned short Wlds[128 * 32];
  const int tid  = threadIdx.x;
  const int wave = tid >> 5, lane = tid & 31;
  const int half = (lane >> 4) & 1, mr = lane & 15;
  const int m0   = blockIdx.x * 128;

  floatx8 c[8] = {};
  for (int k0 = 0; k0 < K; k0 += 32) {
    for (int e4 = tid * 4; e4 < 4096; e4 += 1024) {
      const int row = e4 >> 5, kk = e4 & 31;        // X: 4 consecutive k
      const float4v xv = *(const float4v*)(X + (size_t)(m0 + row) * ldx + k0 + kk);
      uint2v xp; xp.x = pack2(xv[0], xv[1]); xp.y = pack2(xv[2], xv[3]);
      *(uint2v*)(Xlds + e4) = xp;
      const int n = e4 & 127, kw = e4 >> 7;         // W: 4 consecutive n
      const float4v wv = *(const float4v*)(W + (size_t)(k0 + kw) * ldw + n);
#pragma unroll
      for (int j = 0; j < 4; ++j) Wlds[(n + j) * 32 + kw] = f2bf(wv[j]);
    }
    __syncthreads();
    FragBF a;
    const unsigned short* arow = Xlds + (wave * 16 + mr) * 32;
    a.q[0] = *(const uint4v*)(arow + half * 8);
    a.q[1] = *(const uint4v*)(arow + 16 + half * 8);
#pragma unroll
    for (int nt = 0; nt < 8; ++nt) {
      FragBF bfr;
      const unsigned short* bcol = Wlds + (nt * 16 + mr) * 32 + half * 8;
      bfr.q[0] = *(const uint4v*)(bcol);
      bfr.q[1] = *(const uint4v*)(bcol + 16);
      c[nt] = __builtin_amdgcn_wmma_f32_16x16x32_bf16(false, a.b, false, bfr.b,
                                                      (short)0, c[nt], false, false);
    }
    __syncthreads();
  }
#pragma unroll
  for (int nt = 0; nt < 8; ++nt) {
    const int n = nt * 16 + mr;
    const float bv = bias ? bias[n] : 0.0f;
#pragma unroll
    for (int r = 0; r < 8; ++r) {
      const int m = m0 + wave * 16 + r + half * 8;
      Y[(size_t)m * ldy + n] = c[nt][r] + bv;
    }
  }
}

// ===================================================== WMMA aggregation
// agg[b,s,h,0:16] = sum_t attn[b,h,s,t] * tp[b,t,h*16:(h+1)*16]
// Block (8 waves) covers 128 s-rows; (b,h) V-tile staged to LDS once.
__global__ __launch_bounds__(256) void agg_wmma(
    const float* __restrict__ attn, const float* __restrict__ tp,
    float* __restrict__ agg, int Ns, int Nt)
{
  extern __shared__ unsigned short smem[];
  unsigned short* Vlds = smem;             // [d][t] : d*Nt + t
  unsigned short* Alds = smem + 16 * Nt;   // [row][kk]
  const int tid  = threadIdx.x;
  const int wave = tid >> 5, lane = tid & 31;
  const int half = (lane >> 4) & 1, mr = lane & 15;
  const int s0 = blockIdx.x * 128, h = blockIdx.y, b = blockIdx.z;

  const float* tpb = tp + (size_t)b * Nt * 128 + h * 16;
  for (int e4 = tid * 4; e4 < Nt * 16; e4 += 1024) {
    const int t = e4 >> 4, d = e4 & 15;              // 4 consecutive d
    const float4v v = *(const float4v*)(tpb + (size_t)t * 128 + d);
#pragma unroll
    for (int j = 0; j < 4; ++j) Vlds[(d + j) * Nt + t] = f2bf(v[j]);
  }
  __syncthreads();

  const float* Xb = attn + ((size_t)(b * 8 + h) * Ns + s0) * Nt;
  floatx8 c = {};
  for (int k0 = 0; k0 < Nt; k0 += 32) {
    for (int e4 = tid * 4; e4 < 4096; e4 += 1024) {
      const int row = e4 >> 5, kk = e4 & 31;
      const float4v xv = *(const float4v*)(Xb + (size_t)row * Nt + k0 + kk);
      uint2v xp; xp.x = pack2(xv[0], xv[1]); xp.y = pack2(xv[2], xv[3]);
      *(uint2v*)(Alds + e4) = xp;
    }
    __syncthreads();
    FragBF a, bfr;
    const unsigned short* arow = Alds + (wave * 16 + mr) * 32;
    a.q[0] = *(const uint4v*)(arow + half * 8);
    a.q[1] = *(const uint4v*)(arow + 16 + half * 8);
    const unsigned short* bcol = Vlds + mr * Nt + k0 + half * 8;
    bfr.q[0] = *(const uint4v*)(bcol);
    bfr.q[1] = *(const uint4v*)(bcol + 16);
    c = __builtin_amdgcn_wmma_f32_16x16x32_bf16(false, a.b, false, bfr.b,
                                                (short)0, c, false, false);
    __syncthreads();
  }
#pragma unroll
  for (int r = 0; r < 8; ++r) {
    const int s = s0 + wave * 16 + r + half * 8;
    agg[(size_t)(b * Ns + s) * 128 + h * 16 + mr] = c[r];
  }
}

// ================================================= per-head score vectors
// TRANSPOSED output: out[h*rows + row] so softmax reads are unit-stride in t.
__global__ void score_kernel(const float* __restrict__ proj,
                             const float* __restrict__ a,
                             float* __restrict__ out, int rows)
{
  const int idx = blockIdx.x * blockDim.x + threadIdx.x;
  if (idx >= rows * 8) return;
  const int row = idx >> 3, h = idx & 7;
  const float* p  = proj + (size_t)row * 128 + h * 16;
  const float* av = a + h * 16;
  float s = 0.f;
#pragma unroll
  for (int j = 0; j < 4; ++j) {
    const float4v pv = *(const float4v*)(p + j * 4);
    const float4v avv = *(const float4v*)(av + j * 4);
#pragma unroll
    for (int d = 0; d < 4; ++d) s += pv[d] * avv[d];
  }
  out[h * rows + row] = s;
}

// ============================================== masked leakyrelu + softmax
// sc_* layout: [h][b*N + n]. 4 t-values per thread: b128 row stores.
__global__ __launch_bounds__(128) void attn_softmax(
    const float* __restrict__ sc_src, const float* __restrict__ sc_trg,
    const unsigned char* __restrict__ smask, const unsigned char* __restrict__ tmask,
    float* __restrict__ attn, int Ns, int Nt, int sRows, int tRows)
{
  __shared__ float buf[1024];
  __shared__ float red[128];
  const int s = blockIdx.x, h = blockIdx.y, b = blockIdx.z;
  const int tid = threadIdx.x;
  float* row = attn + ((size_t)(b * 8 + h) * Ns + s) * Nt;

  if (!smask[b * Ns + s]) {                      // row_valid == false -> zeros
    const float4v z = {0.f, 0.f, 0.f, 0.f};
    for (int t4 = tid * 4; t4 < Nt; t4 += 512) *(float4v*)(row + t4) = z;
    return;
  }
  const float ss = sc_src[h * sRows + b * Ns + s];
  const float* st = sc_trg + h * tRows + b * Nt;
  const unsigned char* tm = tmask + b * Nt;

  float lmax = -3.4028235e38f;
  for (int t4 = tid * 4; t4 < Nt; t4 += 512) {
    const float4v sv = *(const float4v*)(st + t4);         // unit-stride b128
    const unsigned m4 = *(const unsigned*)(tm + t4);       // 4 mask bytes
    float4v o;
#pragma unroll
    for (int j = 0; j < 4; ++j) {
      float v = ss + sv[j];
      v = (v > 0.0f) ? v : 0.2f * v;                       // leaky_relu(0.2)
      if (!((m4 >> (8 * j)) & 0xFF)) v = -3.4028235e38f;   // finfo.min
      o[j] = v;
      lmax = fmaxf(lmax, v);
    }
    *(float4v*)(buf + t4) = o;
  }
  red[tid] = lmax; __syncthreads();
  for (int w = 64; w > 0; w >>= 1) {
    if (tid < w) red[tid] = fmaxf(red[tid], red[tid + w]);
    __syncthreads();
  }
  const float rmax = red[0]; __syncthreads();

  float lsum = 0.f;
  for (int t4 = tid * 4; t4 < Nt; t4 += 512) {
    float4v v = *(const float4v*)(buf + t4);
#pragma unroll
    for (int j = 0; j < 4; ++j) { v[j] = expf(v[j] - rmax); lsum += v[j]; }
    *(float4v*)(buf + t4) = v;
  }
  red[tid] = lsum; __syncthreads();
  for (int w = 64; w > 0; w >>= 1) {
    if (tid < w) red[tid] += red[tid + w];
    __syncthreads();
  }
  const float inv = 1.0f / red[0];
  for (int t4 = tid * 4; t4 < Nt; t4 += 512) {
    float4v v = *(const float4v*)(buf + t4);
#pragma unroll
    for (int j = 0; j < 4; ++j) v[j] *= inv;
    *(float4v*)(row + t4) = v;                             // b128 stream out
  }
}

// ===================================== LN(1e-5) -> GELU -> +res -> LN(1e-6)
__device__ inline float block_sum(float v, float* red, int tid) {
  red[tid] = v; __syncthreads();
  for (int w = 64; w > 0; w >>= 1) {
    if (tid < w) red[tid] += red[tid + w];
    __syncthreads();
  }
  const float r = red[0]; __syncthreads();
  return r;
}

__global__ __launch_bounds__(128) void fuse_finish(
    const float* __restrict__ h, const float* __restrict__ feat,
    const float* __restrict__ lfw, const float* __restrict__ lfb,
    const float* __restrict__ low, const float* __restrict__ lob,
    float* __restrict__ out)
{
  __shared__ float red[128];
  const int row = blockIdx.x, tid = threadIdx.x;
  float x = h[(size_t)row * 128 + tid];

  float m  = block_sum(x, red, tid) * (1.0f / 128.0f);
  float ms = block_sum(x * x, red, tid) * (1.0f / 128.0f);
  float xn = (x - m) * rsqrtf(ms - m * m + 1e-5f) * lfw[tid] + lfb[tid];
  float g  = 0.5f * xn * (1.0f + erff(xn * 0.70710678118654752f));
  float y  = g + feat[(size_t)row * 128 + tid];

  m  = block_sum(y, red, tid) * (1.0f / 128.0f);
  ms = block_sum(y * y, red, tid) * (1.0f / 128.0f);
  out[(size_t)row * 128 + tid] = (y - m) * rsqrtf(ms - m * m + 1e-6f) * low[tid] + lob[tid];
}

// ================================================================ driver
extern "C" void kernel_launch(void* const* d_in, const int* in_sizes, int n_in,
                              void* d_out, int out_size, void* d_ws, size_t ws_size,
                              hipStream_t stream) {
  constexpr int B = 4, NS = 1024, NT = 512, H = 8;
  constexpr int MM = B * NS;   // 4096 meas rows
  constexpr int MT = B * NT;   // 2048 target rows

  const float* meas = (const float*)d_in[0];
  const float* targ = (const float*)d_in[1];
  const unsigned char* mmask = (const unsigned char*)d_in[2];
  const unsigned char* tmask = (const unsigned char*)d_in[3];
  const float* W_src_mm = (const float*)d_in[4];
  const float* a_src_mm = (const float*)d_in[5];
  const float* a_trg_mm = (const float*)d_in[6];
  const float* W_out_mm = (const float*)d_in[7];
  const float* W_src_ms = (const float*)d_in[8];
  const float* W_trg_ms = (const float*)d_in[9];
  const float* a_src_ms = (const float*)d_in[10];
  const float* a_trg_ms = (const float*)d_in[11];
  const float* W_out_ms = (const float*)d_in[12];
  const float* W_src_sm = (const float*)d_in[13];
  const float* W_trg_sm = (const float*)d_in[14];
  const float* a_src_sm = (const float*)d_in[15];
  const float* a_trg_sm = (const float*)d_in[16];
  const float* W_out_sm = (const float*)d_in[17];
  const float* W_src_ss = (const float*)d_in[18];
  const float* a_src_ss = (const float*)d_in[19];
  const float* a_trg_ss = (const float*)d_in[20];
  const float* W_out_ss = (const float*)d_in[21];
  const float* W_fuse_meas = (const float*)d_in[22];
  const float* b_fuse_meas = (const float*)d_in[23];
  const float* lf_m_w = (const float*)d_in[24];
  const float* lf_m_b = (const float*)d_in[25];
  const float* lo_m_w = (const float*)d_in[26];
  const float* lo_m_b = (const float*)d_in[27];
  const float* W_fuse_targ = (const float*)d_in[28];
  const float* b_fuse_targ = (const float*)d_in[29];
  const float* lf_t_w = (const float*)d_in[30];
  const float* lf_t_b = (const float*)d_in[31];
  const float* lo_t_w = (const float*)d_in[32];
  const float* lo_t_b = (const float*)d_in[33];

  // d_out layout (flat float32, return order)
  float* out = (float*)d_out;
  float* out_meas = out;                                  // [4,1024,128]
  float* out_targ = out_meas + (size_t)MM * 128;          // [4,512,128]
  float* attn_mm  = out_targ + (size_t)MT * 128;          // [4,8,1024,1024]
  float* attn_ms  = attn_mm + (size_t)B * H * NS * NS;    // [4,8,1024,512]
  float* attn_sm  = attn_ms + (size_t)B * H * NS * NT;    // [4,8,512,1024]
  float* attn_ss  = attn_sm + (size_t)B * H * NT * NS;    // [4,8,512,512]

  // workspace (bump allocator over d_ws)
  float* ws = (float*)d_ws;
  size_t off = 0;
  auto alloc = [&](size_t n) { float* p = ws + off; off += n; return p; };
  float* sp_mm = alloc((size_t)MM * 128);
  float* sp_ms = alloc((size_t)MM * 128);
  float* tp_ms = alloc((size_t)MT * 128);
  float* sp_sm = alloc((size_t)MT * 128);
  float* tp_sm = alloc((size_t)MM * 128);
  float* sp_ss = alloc((size_t)MT * 128);
  float* sc_src_mm = alloc((size_t)MM * H);
  float* sc_trg_mm = alloc((size_t)MM * H);
  float* sc_src_ms = alloc((size_t)MM * H);
  float* sc_trg_ms = alloc((size_t)MT * H);
  float* sc_src_sm = alloc((size_t)MT * H);
  float* sc_trg_sm = alloc((size_t)MM * H);
  float* sc_src_ss = alloc((size_t)MT * H);
  float* sc_trg_ss = alloc((size_t)MT * H);
  float* agg_mm = alloc((size_t)MM * 128);
  float* agg_ms = alloc((size_t)MM * 128);
  float* agg_sm = alloc((size_t)MT * 128);
  float* agg_ss = alloc((size_t)MT * 128);
  float* fin_meas = alloc((size_t)MM * 256);
  float* fin_targ = alloc((size_t)MT * 256);
  float* hf_meas  = alloc((size_t)MM * 128);
  float* hf_targ  = alloc((size_t)MT * 128);

  const dim3 blk(256);
  const dim3 gM(MM / 128), gT(MT / 128);

  // 1) projections (bf16 WMMA, fp32 accumulate)
  gemm_wmma<<<gM, blk, 0, stream>>>(meas, W_src_mm, nullptr, sp_mm, 128, 128, 128, 128);
  gemm_wmma<<<gM, blk, 0, stream>>>(meas, W_src_ms, nullptr, sp_ms, 128, 128, 128, 128);
  gemm_wmma<<<gT, blk, 0, stream>>>(targ, W_trg_ms, nullptr, tp_ms, 128, 128, 128, 128);
  gemm_wmma<<<gT, blk, 0, stream>>>(targ, W_src_sm, nullptr, sp_sm, 128, 128, 128, 128);
  gemm_wmma<<<gM, blk, 0, stream>>>(meas, W_trg_sm, nullptr, tp_sm, 128, 128, 128, 128);
  gemm_wmma<<<gT, blk, 0, stream>>>(targ, W_src_ss, nullptr, sp_ss, 128, 128, 128, 128);

  // 2) per-head attention score vectors ([h][row] layout)
  const int tM = (MM * H + 255) / 256, tT = (MT * H + 255) / 256;
  score_kernel<<<tM, 256, 0, stream>>>(sp_mm, a_src_mm, sc_src_mm, MM);
  score_kernel<<<tM, 256, 0, stream>>>(sp_mm, a_trg_mm, sc_trg_mm, MM);
  score_kernel<<<tM, 256, 0, stream>>>(sp_ms, a_src_ms, sc_src_ms, MM);
  score_kernel<<<tT, 256, 0, stream>>>(tp_ms, a_trg_ms, sc_trg_ms, MT);
  score_kernel<<<tT, 256, 0, stream>>>(sp_sm, a_src_sm, sc_src_sm, MT);
  score_kernel<<<tM, 256, 0, stream>>>(tp_sm, a_trg_sm, sc_trg_sm, MM);
  score_kernel<<<tT, 256, 0, stream>>>(sp_ss, a_src_ss, sc_src_ss, MT);
  score_kernel<<<tT, 256, 0, stream>>>(sp_ss, a_trg_ss, sc_trg_ss, MT);

  // 3) masked leaky-relu softmax -> attn tensors (b128 streaming writes)
  attn_softmax<<<dim3(NS, H, B), 128, 0, stream>>>(sc_src_mm, sc_trg_mm, mmask, mmask, attn_mm, NS, NS, MM, MM);
  attn_softmax<<<dim3(NS, H, B), 128, 0, stream>>>(sc_src_ms, sc_trg_ms, mmask, tmask, attn_ms, NS, NT, MM, MT);
  attn_softmax<<<dim3(NT, H, B), 128, 0, stream>>>(sc_src_sm, sc_trg_sm, tmask, mmask, attn_sm, NT, NS, MT, MM);
  attn_softmax<<<dim3(NT, H, B), 128, 0, stream>>>(sc_src_ss, sc_trg_ss, tmask, tmask, attn_ss, NT, NT, MT, MT);

  // 4) attn @ V per head (WMMA; V-tile LDS-resident per block)
  const size_t smem_mm = (size_t)(16 * NS + 128 * 32) * 2;  // Nt = NS
  const size_t smem_ms = (size_t)(16 * NT + 128 * 32) * 2;  // Nt = NT
  agg_wmma<<<dim3(NS / 128, H, B), blk, smem_mm, stream>>>(attn_mm, sp_mm, agg_mm, NS, NS);
  agg_wmma<<<dim3(NS / 128, H, B), blk, smem_ms, stream>>>(attn_ms, tp_ms, agg_ms, NS, NT);
  agg_wmma<<<dim3(NT / 128, H, B), blk, smem_mm, stream>>>(attn_sm, tp_sm, agg_sm, NT, NS);
  agg_wmma<<<dim3(NT / 128, H, B), blk, smem_ms, stream>>>(attn_ss, sp_ss, agg_ss, NT, NT);

  // 5) out-projections fused with concat: interleaved column halves (ldy=256)
  gemm_wmma<<<gM, blk, 0, stream>>>(agg_mm, W_out_mm, nullptr, fin_meas,       128, 128, 128, 256);
  gemm_wmma<<<gM, blk, 0, stream>>>(agg_ms, W_out_ms, nullptr, fin_meas + 128, 128, 128, 128, 256);
  gemm_wmma<<<gT, blk, 0, stream>>>(agg_ss, W_out_ss, nullptr, fin_targ,       128, 128, 128, 256);
  gemm_wmma<<<gT, blk, 0, stream>>>(agg_sm, W_out_sm, nullptr, fin_targ + 128, 128, 128, 128, 256);

  // 6) fuse GEMM (K=256) + bias
  gemm_wmma<<<gM, blk, 0, stream>>>(fin_meas, W_fuse_meas, b_fuse_meas, hf_meas, 256, 256, 128, 128);
  gemm_wmma<<<gT, blk, 0, stream>>>(fin_targ, W_fuse_targ, b_fuse_targ, hf_targ, 256, 256, 128, 128);

  // 7) LN -> GELU -> residual -> LN
  fuse_finish<<<MM, 128, 0, stream>>>(hf_meas, meas, lf_m_w, lf_m_b, lo_m_w, lo_m_b, out_meas);
  fuse_finish<<<MT, 128, 0, stream>>>(hf_targ, targ, lf_t_w, lf_t_b, lo_t_w, lo_t_b, out_targ);
}
